// DynamicConv_8211977470730
// MI455X (gfx1250) — matrix operand order repaired and checked by reading, not compiled, not run on hardware
//
#include <hip/hip_runtime.h>

typedef __attribute__((ext_vector_type(16))) __bf16 v16bf;
typedef __attribute__((ext_vector_type(8)))  __bf16 v8bf;
typedef __attribute__((ext_vector_type(8)))  float  v8f;

#define BATCH   16
#define C_IN    96
#define F_DIM   64
#define T_DIM   500
#define C_OUT   96
#define KSZ     3
#define KCAND   4
#define CTX     128
#define TEMP    30.0f

#define T_TILE  96           // t positions per block (3 subtiles of 32)
#define F_TILE  4            // f rows per block
#define N_SUB   2            // N-tiles (16 wide) per wave -> N=32
#define NTHREADS 384         // 12 waves = 4 f x 3 t-subtiles
#define TLOC    (T_TILE + 2) // 98: t positions incl. halo
#define PLANES  (F_TILE + 2) // 6: f rows incl. halo
#define CI_PAD  104          // padded ci stride in LDS (halves); rows 16B aligned
#define M_TILES 6            // 96 / 16 output-channel tiles
#define NFRAG   (BATCH * 9 * 3 * M_TILES)   // 2592 A fragments
#define FRAG_ELEMS 512                       // 16x32 bf16 per fragment
#define A_HALVES   (9 * 3 * M_TILES * FRAG_ELEMS)  // 82944 bf16 per batch
#define X_HALVES   (PLANES * TLOC * CI_PAD)        // 61152 bf16

// ---------------------------------------------------------------------------
// Kernel 0: attention softmax  (B x K tiny GEMV + softmax)
// ---------------------------------------------------------------------------
__global__ void att_kernel(const float* __restrict__ g,
                           const float* __restrict__ Wr,
                           const float* __restrict__ br,
                           float* __restrict__ att) {
    int b = threadIdx.x;
    if (b >= BATCH) return;
    float l[KCAND];
    float mx = -1e30f;
    for (int k = 0; k < KCAND; ++k) {
        float s = br[k];
        const float* gv = g + b * CTX;
        const float* wv = Wr + k * CTX;
        for (int j = 0; j < CTX; ++j) s += gv[j] * wv[j];
        s *= (1.0f / TEMP);
        l[k] = s;
        mx = fmaxf(mx, s);
    }
    float den = 0.f;
    for (int k = 0; k < KCAND; ++k) { l[k] = __expf(l[k] - mx); den += l[k]; }
    float inv = 1.0f / den;
    for (int k = 0; k < KCAND; ++k) att[b * KCAND + k] = l[k] * inv;
}

// ---------------------------------------------------------------------------
// Kernel 1: mix candidate filters with attention and emit bf16 A-fragments
// in exact v_wmma 16x16x32 A-operand lane layout.
// Fragment order (off-major, so each (df,dt) group is contiguous in LDS):
//   frag index = ((b*9 + off)*3 + kt)*6 + m ; element = lane*16 + h
// ---------------------------------------------------------------------------
__global__ void frag_kernel(const float* __restrict__ W,
                            const float* __restrict__ att,
                            __bf16* __restrict__ wsA) {
    int gidx = blockIdx.x * 256 + threadIdx.x;
    const int TOTAL = NFRAG * FRAG_ELEMS;
    if (gidx >= TOTAL) return;

    int frag   = gidx >> 9;
    int within = gidx & 511;
    int lane   = within >> 4;
    int h      = within & 15;

    int m   = frag % 6;  int t1 = frag / 6;
    int kt  = t1 % 3;    t1 /= 3;
    int off = t1 % 9;    int b  = t1 / 9;

    // (lane, h) -> (row m, k) per CDNA5 16-bit A-matrix 16x32 layout
    int v  = h >> 1;
    int j  = h & 1;
    int mm = lane & 15;
    int hi = lane >> 4;
    int kk = (v < 4) ? (hi * 8 + 2 * v + j)
                     : (16 + hi * 8 + 2 * (v - 4) + j);

    int co = m * 16 + mm;
    int ci = kt * 32 + kk;
    int wflat = (co * C_IN + ci) * (KSZ * KSZ) + off;   // off = df*3+dt

    const int D = C_OUT * C_IN * KSZ * KSZ;             // 82944
    const float* a = att + b * KCAND;
    float s = a[0] * W[0 * D + wflat] + a[1] * W[1 * D + wflat]
            + a[2] * W[2 * D + wflat] + a[3] * W[3 * D + wflat];
    wsA[gidx] = (__bf16)s;
}

// ---------------------------------------------------------------------------
// Kernel 2: implicit-GEMM dynamic conv, bf16 WMMA (f32 accum), LDS-fed.
// Block = 4 f-rows x 96 t, 12 waves = 4 f x 3 t-subtiles; each wave computes
// a 96(co) x 32(t) output tile (12 accumulators). 3 waves/SIMD occupancy,
// x halo amplification 1.5x  ->  ~491 MB HBM total (~21 us floor @23.3TB/s).
// LDS: [A fragments for batch b : 165,888 B][x halo tile : 122,304 B]
// ---------------------------------------------------------------------------
__global__ __launch_bounds__(NTHREADS)
__attribute__((amdgpu_waves_per_eu(1, 3)))
void conv_kernel(const float* __restrict__ x,
                 const __bf16* __restrict__ wsA,
                 float* __restrict__ out) {
    extern __shared__ __bf16 lds[];
    __bf16* ldsA = lds;                 // A_HALVES
    __bf16* ldsX = lds + A_HALVES;      // X_HALVES, [plane][tl][ci]

    const int t0  = blockIdx.x * T_TILE;
    const int f0  = blockIdx.y * F_TILE;
    const int b   = blockIdx.z;
    const int tid = threadIdx.x;

    // ---- stage per-batch A fragments (L2-resident) into LDS ----
    {
        const __bf16* gA = wsA + (size_t)b * A_HALVES;
        const int CH = A_HALVES / 8;    // 10368 16-byte chunks
        for (int i = tid; i < CH; i += NTHREADS)
            *(v8bf*)(ldsA + i * 8) = *(const v8bf*)(gA + i * 8);
    }

    // ---- stage x halo tile (6 f-rows x 98 t x 96 ci) as bf16, t-major ----
    {
        const int NL = PLANES * TLOC * C_IN;   // 56448
        for (int idx = tid; idx < NL; idx += NTHREADS) {
            int tl = idx % TLOC;
            int r  = idx / TLOC;
            int ci = r % C_IN;
            int p  = r / C_IN;                 // plane 0..5
            int fr = f0 + p - 1;
            int tg = t0 + tl - 1;
            float val = 0.f;
            if (fr >= 0 && fr < F_DIM && tg >= 0 && tg < T_DIM)
                val = x[(((size_t)b * C_IN + ci) * F_DIM + fr) * T_DIM + tg];
            ldsX[(p * TLOC + tl) * CI_PAD + ci] = (__bf16)val;
        }
    }
    __syncthreads();

    const int w    = tid >> 5;       // wave id 0..11
    const int fsub = w / 3;          // 0..3 : which f row
    const int tw   = w % 3;          // 0..2 : which 32-wide t subtile
    const int lane = tid & 31;
    const int n    = lane & 15;      // N column (t offset)
    const int hi   = lane >> 4;      // K half for B / M half for C

    v8f acc[M_TILES * N_SUB];
    for (int m = 0; m < M_TILES * N_SUB; ++m)
        acc[m] = (v8f){0.f,0.f,0.f,0.f,0.f,0.f,0.f,0.f};

    const int laneAoff = lane * 16;

    for (int df = 0; df < 3; ++df) {
        const int plane = fsub + df;
        for (int dt = 0; dt < 3; ++dt) {
            const int off = df * 3 + dt;
            const int tlBase = tw * 32 + n + dt;        // + j*16, 0..97
            const __bf16* prow = ldsX + ((size_t)plane * TLOC + tlBase) * CI_PAD
                               + hi * 8;
            // contiguous 18 KB group: [kt][m] fragments, imm-offset reachable
            const __bf16* abase = ldsA + (size_t)off * (3 * M_TILES * FRAG_ELEMS)
                                + laneAoff;

            #pragma unroll
            for (int kt = 0; kt < 3; ++kt) {
                // two B fragments: 32(ci) x 16(t) each; lane K = hi*8 + 0..7, 16 + hi*8 + 0..7
                union { v16bf v; v8bf h[2]; } bu[N_SUB];
                #pragma unroll
                for (int j = 0; j < N_SUB; ++j) {
                    const __bf16* bp = prow + (size_t)j * 16 * CI_PAD + kt * 32;
                    bu[j].h[0] = *(const v8bf*)(bp);
                    bu[j].h[1] = *(const v8bf*)(bp + 16);
                }

                // preload all 6 A fragments (imm offsets from abase)
                union { v16bf v; v8bf h[2]; } au[M_TILES];
                #pragma unroll
                for (int m = 0; m < M_TILES; ++m) {
                    const __bf16* ap = abase + ((size_t)kt * M_TILES + m) * FRAG_ELEMS;
                    au[m].h[0] = *(const v8bf*)(ap);
                    au[m].h[1] = *(const v8bf*)(ap + 8);
                }

                // 12 back-to-back WMMAs: each A fragment feeds both N-tiles
                #pragma unroll
                for (int m = 0; m < M_TILES; ++m) {
                    #pragma unroll
                    for (int j = 0; j < N_SUB; ++j) {
                        acc[m * N_SUB + j] = __builtin_amdgcn_wmma_f32_16x16x32_bf16(
                            false, au[m].v, false, bu[j].v, (short)0,
                            acc[m * N_SUB + j], false, false);
                    }
                }
            }
        }
    }

    // ---- epilogue: C/D layout -> out[b][co][f][t] ----
    const int f = f0 + fsub;
    for (int j = 0; j < N_SUB; ++j) {
        const int t = t0 + tw * 32 + j * 16 + n;
        if (t < T_DIM) {
            for (int m = 0; m < M_TILES; ++m) {
                const int coBase = m * 16 + hi * 8;
                for (int r = 0; r < 8; ++r) {
                    out[(((size_t)b * C_OUT + coBase + r) * F_DIM + f) * T_DIM + t]
                        = acc[m * N_SUB + j][r];
                }
            }
        }
    }
}

// ---------------------------------------------------------------------------
extern "C" void kernel_launch(void* const* d_in, const int* in_sizes, int n_in,
                              void* d_out, int out_size, void* d_ws, size_t ws_size,
                              hipStream_t stream) {
    const float* x  = (const float*)d_in[0];
    const float* g  = (const float*)d_in[1];
    const float* Wr = (const float*)d_in[2];
    const float* br = (const float*)d_in[3];
    const float* W  = (const float*)d_in[4];

    float*  att = (float*)d_ws;                       // 64 floats
    __bf16* wsA = (__bf16*)((char*)d_ws + 256);       // 2592 * 512 bf16

    att_kernel<<<1, 32, 0, stream>>>(g, Wr, br, att);

    frag_kernel<<<(NFRAG * FRAG_ELEMS) / 256, 256, 0, stream>>>(W, att, wsA);

    dim3 grid((T_DIM + T_TILE - 1) / T_TILE, F_DIM / F_TILE, BATCH);
    size_t shbytes = (size_t)(A_HALVES + X_HALVES) * sizeof(__bf16); // 288,192 B
    conv_kernel<<<grid, NTHREADS, shbytes, stream>>>(x, wsA, (float*)d_out);
}
